// Net_86749749445064
// MI455X (gfx1250) — matrix operand order repaired and checked by reading, not compiled, not run on hardware
//
#include <hip/hip_runtime.h>
#include <hip/hip_bf16.h>

// ---------------------------------------------------------------------------
// GMT-style GNN forward for MI455X (gfx1250, wave32, WMMA).
// Dense matmuls: v_wmma_f32_16x16x32_f16, f16 inputs / f32 accumulate.
// gemm_fast caches all weight fragments in registers (K compile-time) and
// grid-strides over M stripes; A fragments are 4x global_load_b128.
// GCN edge passes are memory/L2-atomic bound: one thread per (edge, 4 chans),
// float4 gather + 4 batched global f32 atomics.
// ---------------------------------------------------------------------------

typedef __attribute__((ext_vector_type(16))) _Float16 v16h;
typedef __attribute__((ext_vector_type(8)))  float    v8f;

#define NNODES 131072   // B*N
#define NEDGES 1048576  // NNODES*DEG
#define BGR    512
#define HDIM   64

// ---- WMMA lane-layout (16x16x32 f16, wave32) ------------------------------
// A (16x32): lane m = lane&15, hl = lane>>4.
//   element j: k = j + 8*hl (j<8) ; k = 16 + (j-8) + 8*hl (j>=8)
// B (32x16): lane n = lane&15, hl = lane>>4; element j: same k as A (k = j + (j&8) + 8*hl)
// C/D (16x16 f32): element i -> row m = i + 8*hl, col n = lane&15.

__device__ __forceinline__ int a_kidx(int j, int hl) {
    int r = j >> 1, bit = j & 1;
    return 16 * (r >> 2) + 2 * (r & 3) + 8 * hl + bit;
}
__device__ __forceinline__ int b_kidx(int j, int hl) {
    return a_kidx(j, hl);
}

// A fragment: rowk0 = &A[row*lda + k0]; requires 16B alignment of rowk0+8*hl.
__device__ __forceinline__ v16h load_a_frag(const float* __restrict__ rowk0, int hl)
{
    const float4* p = reinterpret_cast<const float4*>(rowk0 + 8 * hl);
    float4 q0 = p[0], q1 = p[1], q2 = p[4], q3 = p[5];
    v16h a;
    a[0]  = (_Float16)q0.x; a[1]  = (_Float16)q0.y;
    a[2]  = (_Float16)q0.z; a[3]  = (_Float16)q0.w;
    a[4]  = (_Float16)q1.x; a[5]  = (_Float16)q1.y;
    a[6]  = (_Float16)q1.z; a[7]  = (_Float16)q1.w;
    a[8]  = (_Float16)q2.x; a[9]  = (_Float16)q2.y;
    a[10] = (_Float16)q2.z; a[11] = (_Float16)q2.w;
    a[12] = (_Float16)q3.x; a[13] = (_Float16)q3.y;
    a[14] = (_Float16)q3.z; a[15] = (_Float16)q3.w;
    return a;
}

// B fragment: colbase = &W[k0*stride + n]; 16 row-strided loads (batched).
__device__ __forceinline__ v16h load_b_frag_strided(const float* __restrict__ colbase,
                                                    long stride, int hl)
{
    const float* p = colbase + (long)8 * hl * stride;
    v16h b;
#pragma unroll
    for (int j = 0; j < 16; ++j) b[j] = (_Float16)p[(long)(j + (j & 8)) * stride];
    return b;
}

// ---------------------------------------------------------------------------
// Fast WMMA GEMM: C[M, NT*16] = act(A[M, KT*32] @ W + bias) (+resid)
// Weight fragments cached in registers; wave grid-strides over 16-row stripes.
// Requires: M%16==0, lda%4==0, 16B-aligned buffers.
// ---------------------------------------------------------------------------
template <int NT, int KT>
__global__ void gemm_fast(const float* __restrict__ A, int lda,
                          const float* __restrict__ W,
                          const float* __restrict__ bias,
                          const float* __restrict__ resid, int ldr,
                          float* __restrict__ Cg, int ldc, int M, int relu)
{
    const int N = NT * 16;
    int lane = threadIdx.x & 31;
    int l15 = lane & 15, hl = lane >> 4;

    v16h bf[KT][NT];
#pragma unroll
    for (int kt = 0; kt < KT; ++kt)
#pragma unroll
        for (int t = 0; t < NT; ++t)
            bf[kt][t] = load_b_frag_strided(W + (long)kt * 32 * N + t * 16 + l15, N, hl);

    float bv = bias ? bias[l15] : 0.f;          // re-fetched per tile column below
    (void)bv;

    for (int m0 = blockIdx.x * 16; m0 < M; m0 += (int)gridDim.x * 16) {
        const float* arow = A + (long)(m0 + l15) * lda;
        v16h af[KT];
#pragma unroll
        for (int kt = 0; kt < KT; ++kt) af[kt] = load_a_frag(arow + kt * 32, hl);

        v8f acc[NT] = {};
#pragma unroll
        for (int kt = 0; kt < KT; ++kt)
#pragma unroll
            for (int t = 0; t < NT; ++t)
                acc[t] = __builtin_amdgcn_wmma_f32_16x16x32_f16(
                    false, af[kt], false, bf[kt][t], (short)0, acc[t], false, false);

#pragma unroll
        for (int t = 0; t < NT; ++t) {
            int n = t * 16 + l15;
            float bb = bias ? bias[n] : 0.f;
#pragma unroll
            for (int i = 0; i < 8; ++i) {
                int m = m0 + i + 8 * hl;
                float v = acc[t][i] + bb;
                if (relu) v = fmaxf(v, 0.f);
                if (resid) v += resid[(long)m * ldr + n];
                Cg[(long)m * ldc + n] = v;
            }
        }
    }
}

// ---------------------------------------------------------------------------
// Guarded WMMA GEMM for odd shapes (K=3, M=1, K=16/N=2). Clamped addresses +
// select keep loads unpredicated so they batch behind one wait.
// ---------------------------------------------------------------------------
__global__ void gemm_guard(const float* __restrict__ A, int lda,
                           const float* __restrict__ W, int ldw,
                           const float* __restrict__ bias,
                           float* __restrict__ Cg, int ldc,
                           int M, int K, int N, int relu)
{
    int m0 = blockIdx.x * 16;
    int n0 = blockIdx.y * 16;
    int lane = threadIdx.x & 31;
    int l15 = lane & 15, hl = lane >> 4;

    v8f acc = {};
    for (int k0 = 0; k0 < K; k0 += 32) {
        v16h a, b;
#pragma unroll
        for (int j = 0; j < 16; ++j) {
            int ka = k0 + a_kidx(j, hl);
            int arow = m0 + l15;
            bool aok = (arow < M) && (ka < K);
            float av = A[aok ? ((long)arow * lda + ka) : 0];
            a[j] = aok ? (_Float16)av : (_Float16)0.f;

            int kb = k0 + b_kidx(j, hl);
            int ncol = n0 + l15;
            bool bok = (kb < K) && (ncol < N);
            float bw = W[bok ? ((long)kb * ldw + ncol) : 0];
            b[j] = bok ? (_Float16)bw : (_Float16)0.f;
        }
        acc = __builtin_amdgcn_wmma_f32_16x16x32_f16(false, a, false, b,
                                                     (short)0, acc, false, false);
    }
    int n = n0 + l15;
    if (n >= N) return;
    float bv = bias ? bias[n] : 0.f;
#pragma unroll
    for (int i = 0; i < 8; ++i) {
        int m = m0 + i + 8 * hl;
        if (m < M) {
            float v = acc[i] + bv;
            if (relu) v = fmaxf(v, 0.f);
            Cg[(long)m * ldc + n] = v;
        }
    }
}

// ---------------------------------------------------------------------------
// Element-wise / GCN edge kernels (memory & L2-atomic bound)
// ---------------------------------------------------------------------------
__global__ void k_fill0(float* __restrict__ p, long n) {
    long i = (long)blockIdx.x * blockDim.x + threadIdx.x;
    if (i < n) p[i] = 0.f;
}

__global__ void k_edge_deg(const int* __restrict__ dst, float* __restrict__ deg, int E) {
    int e = blockIdx.x * blockDim.x + threadIdx.x;
    if (e < E) atomicAdd(&deg[dst[e]], 1.0f);
}

__global__ void k_make_dis(float* __restrict__ d, int n) { // deg -> rsqrt(deg+1)
    int i = blockIdx.x * blockDim.x + threadIdx.x;
    if (i < n) d[i] = rsqrtf(d[i] + 1.0f);
}

// One thread per (edge, group of 4 channels): float4 gather + 4 atomics.
template <int C>
__global__ void k_gcn_edge(const int* __restrict__ src, const int* __restrict__ dst,
                           const float* __restrict__ dis,
                           const float* __restrict__ hW, float* __restrict__ agg, int E)
{
    const int G = C / 4;
    long t = (long)blockIdx.x * blockDim.x + threadIdx.x;
    if (t >= (long)E * G) return;
    int e = (int)(t / G);
    int g = (int)(t % G);
    int s = src[e], d = dst[e];
    float w = dis[s] * dis[d];
    float4 v = *reinterpret_cast<const float4*>(&hW[(long)s * C + g * 4]);
    float* ap = &agg[(long)d * C + g * 4];
    atomicAdd(ap + 0, v.x * w);
    atomicAdd(ap + 1, v.y * w);
    atomicAdd(ap + 2, v.z * w);
    atomicAdd(ap + 3, v.w * w);
}

template <int C>
__global__ void k_gcn_combine(const float* __restrict__ hW, const float* __restrict__ agg,
                              const float* __restrict__ dis, const float* __restrict__ bias,
                              float* __restrict__ out, int ldo, int n_nodes, int relu)
{
    const int G = C / 4;
    long t = (long)blockIdx.x * blockDim.x + threadIdx.x;
    if (t >= (long)n_nodes * G) return;
    int n = (int)(t / G);
    int g = (int)(t % G);
    float di = dis[n];
    float dd = di * di;
    float4 hv = *reinterpret_cast<const float4*>(&hW[(long)n * C + g * 4]);
    float4 av = *reinterpret_cast<const float4*>(&agg[(long)n * C + g * 4]);
    float4 bv = *reinterpret_cast<const float4*>(&bias[g * 4]);
    float4 o;
    o.x = av.x + hv.x * dd + bv.x;
    o.y = av.y + hv.y * dd + bv.y;
    o.z = av.z + hv.z * dd + bv.z;
    o.w = av.w + hv.w * dd + bv.w;
    if (relu) {
        o.x = fmaxf(o.x, 0.f); o.y = fmaxf(o.y, 0.f);
        o.z = fmaxf(o.z, 0.f); o.w = fmaxf(o.w, 0.f);
    }
    *reinterpret_cast<float4*>(&out[(long)n * ldo + g * 4]) = o;
}

// ---------------------------------------------------------------------------
// Attention (one head per launch), DH=16 zero-padded to K=32.
//  - A fragment: valid only for j<8 (compile-time), 2x float4 from Q.
//  - B fragment: valid only for j<8 (compile-time), 2x float4 from K.
// ---------------------------------------------------------------------------
__global__ void k_attn_scores(const float* __restrict__ Q, long qstride, // 0 => broadcast
                              const float* __restrict__ Kmat,            // [B, Mk, 64]
                              float* __restrict__ scores,                // [B, Sr, Mk]
                              int S, int Sr, int Mk, int head)
{
    int b = blockIdx.x, st = blockIdx.y, mt = blockIdx.z;
    int lane = threadIdx.x & 31;
    int l15 = lane & 15, hl = lane >> 4;
    const float* Qb = Q + (long)b * qstride;
    const float* Kb = Kmat + (long)b * Mk * HDIM;

    int s = st * 16 + l15;
    int sc = (s < S) ? s : 0; // clamp; rows >= S zeroed below
    const float4* qp = reinterpret_cast<const float4*>(Qb + (long)sc * HDIM + head * 16 + 8 * hl);
    float4 q0 = qp[0], q1 = qp[1];
    v16h a = {};
    if (s < S) {
        a[0] = (_Float16)q0.x; a[1] = (_Float16)q0.y;
        a[2] = (_Float16)q0.z; a[3] = (_Float16)q0.w;
        a[4] = (_Float16)q1.x; a[5] = (_Float16)q1.y;
        a[6] = (_Float16)q1.z; a[7] = (_Float16)q1.w;
    }
    v16h bm = {};
    {
        const float4* kp = reinterpret_cast<const float4*>(
            Kb + (long)(mt * 16 + l15) * HDIM + head * 16 + 8 * hl);
        float4 k0 = kp[0], k1 = kp[1];
        bm[0]  = (_Float16)k0.x; bm[1]  = (_Float16)k0.y;
        bm[2]  = (_Float16)k0.z; bm[3]  = (_Float16)k0.w;
        bm[4]  = (_Float16)k1.x; bm[5]  = (_Float16)k1.y;
        bm[6]  = (_Float16)k1.z; bm[7]  = (_Float16)k1.w;
    }
    v8f c = {};
    c = __builtin_amdgcn_wmma_f32_16x16x32_f16(false, a, false, bm, (short)0, c, false, false);

    const float inv_scale = 0.125f; // 1/sqrt(H=64)
#pragma unroll
    for (int i = 0; i < 8; ++i) {
        int sr = st * 16 + i + 8 * hl;
        scores[((long)b * Sr + sr) * Mk + mt * 16 + l15] = c[i] * inv_scale;
    }
}

__global__ void k_attn_softmax(float* __restrict__ scores, int Sr, int Mk)
{
    // grid: (B, S); one wave per valid row; Mk % 32 == 0
    int b = blockIdx.x, s = blockIdx.y;
    float* row = scores + ((long)b * Sr + s) * Mk;
    int lane = threadIdx.x & 31;

    float mx = -1e30f;
    for (int i = lane; i < Mk; i += 32) mx = fmaxf(mx, row[i]);
    for (int off = 16; off > 0; off >>= 1) mx = fmaxf(mx, __shfl_xor(mx, off, 32));

    float sum = 0.f;
    for (int i = lane; i < Mk; i += 32) { float e = __expf(row[i] - mx); row[i] = e; sum += e; }
    for (int off = 16; off > 0; off >>= 1) sum += __shfl_xor(sum, off, 32);

    float inv = 1.f / sum;
    for (int i = lane; i < Mk; i += 32) row[i] *= inv;
}

__global__ void k_attn_av(const float* __restrict__ scores, // [B, Sr, Mk] probs
                          const float* __restrict__ Vmat,   // [B, Mk, 64]
                          const float* __restrict__ Q, long qstride,
                          float* __restrict__ opre,         // [B, S, 64]
                          int S, int Sr, int Mk, int head)
{
    int b = blockIdx.x, st = blockIdx.y;
    int lane = threadIdx.x & 31;
    int l15 = lane & 15, hl = lane >> 4;
    const float* Pb = scores + (long)b * Sr * Mk + (long)(st * 16 + l15) * Mk;
    const float* Vb = Vmat + (long)b * Mk * HDIM;

    v8f acc = {};
    for (int k0 = 0; k0 < Mk; k0 += 32) {
        v16h a = load_a_frag(Pb + k0, hl);
        v16h bm = load_b_frag_strided(Vb + (long)k0 * HDIM + head * 16 + l15, HDIM, hl);
        acc = __builtin_amdgcn_wmma_f32_16x16x32_f16(false, a, false, bm,
                                                     (short)0, acc, false, false);
    }
    const float* Qb = Q + (long)b * qstride;
#pragma unroll
    for (int i = 0; i < 8; ++i) {
        int s = st * 16 + i + 8 * hl;
        if (s < S)
            opre[((long)b * S + s) * HDIM + head * 16 + l15] =
                acc[i] + Qb[(long)s * HDIM + head * 16 + l15];
    }
}

__global__ void k_log_softmax2(const float* __restrict__ logits, float* __restrict__ out, int Bt)
{
    int b = blockIdx.x * blockDim.x + threadIdx.x;
    if (b >= Bt) return;
    float a = logits[b * 2 + 0], c = logits[b * 2 + 1];
    float m = fmaxf(a, c);
    float lse = m + logf(__expf(a - m) + __expf(c - m));
    out[b * 2 + 0] = a - lse;
    out[b * 2 + 1] = c - lse;
}

// ---------------------------------------------------------------------------
// Host driver
// ---------------------------------------------------------------------------
static inline int grid_tiles(int M) { int t = M / 16; return t < 2048 ? t : 2048; }

extern "C" void kernel_launch(void* const* d_in, const int* in_sizes, int n_in,
                              void* d_out, int out_size, void* d_ws, size_t ws_size,
                              hipStream_t stream)
{
    (void)in_sizes; (void)n_in; (void)out_size; (void)ws_size;

    // ---- inputs (flattened in setup_inputs() dict insertion order) --------
    const float* x0  = (const float*)d_in[0];           // [NN,3]
    const int*   ei  = (const int*)  d_in[1];           // [2,E]
    const int*   src = ei;
    const int*   dst = ei + NEDGES;
    // d_in[2] = batch (unused; equal-sized graphs)
    const float* conv1_w = (const float*)d_in[3];
    const float* conv1_b = (const float*)d_in[4];
    const float* conv2_w = (const float*)d_in[5];
    const float* conv2_b = (const float*)d_in[6];
    const float* conv3_w = (const float*)d_in[7];
    const float* conv3_b = (const float*)d_in[8];
    const float* plin1_w = (const float*)d_in[9];
    const float* plin1_b = (const float*)d_in[10];
    const float* S1      = (const float*)d_in[11];      // [64,64]
    const float* m1_qw = (const float*)d_in[12]; const float* m1_qb = (const float*)d_in[13];
    const float* m1_kw = (const float*)d_in[14]; const float* m1_kb = (const float*)d_in[15];
    const float* m1_vw = (const float*)d_in[16]; const float* m1_vb = (const float*)d_in[17];
    const float* m1_ow = (const float*)d_in[18]; const float* m1_ob = (const float*)d_in[19];
    const float* m2_qw = (const float*)d_in[20]; const float* m2_qb = (const float*)d_in[21];
    const float* m2_kw = (const float*)d_in[22]; const float* m2_kb = (const float*)d_in[23];
    const float* m2_vw = (const float*)d_in[24]; const float* m2_vb = (const float*)d_in[25];
    const float* m2_ow = (const float*)d_in[26]; const float* m2_ob = (const float*)d_in[27];
    const float* S3    = (const float*)d_in[28];        // [1,64]
    const float* m3_qw = (const float*)d_in[29]; const float* m3_qb = (const float*)d_in[30];
    const float* m3_kw = (const float*)d_in[31]; const float* m3_kb = (const float*)d_in[32];
    const float* m3_vw = (const float*)d_in[33]; const float* m3_vb = (const float*)d_in[34];
    const float* m3_ow = (const float*)d_in[35]; const float* m3_ob = (const float*)d_in[36];
    const float* plin2_w = (const float*)d_in[37]; const float* plin2_b = (const float*)d_in[38];
    const float* h1_w    = (const float*)d_in[39]; const float* h1_b    = (const float*)d_in[40];
    const float* h2_w    = (const float*)d_in[41]; const float* h2_b    = (const float*)d_in[42];

    float* ws = (float*)d_ws;
    const long NN = NNODES;
    // ---- workspace layout (floats); aliased regions noted ------------------
    const long o_dis  = 0;                    // NN
    const long o_xcat = o_dis  + NN;          // NN*96 (x1|x2|x3 concat)
    const long o_hW   = o_xcat + NN * 96;     // NN*64  (GEMM scratch; later: scores)
    const long o_agg  = o_hW   + NN * 64;     // NN*64  (scatter scratch; later: opre)
    const long o_h    = o_agg  + NN * 64;     // NN*64
    const long o_kd   = o_h    + NN * 64;     // NN*64  (later: k3)
    const long o_vd   = o_kd   + NN * 64;     // NN*64  (later: v3)
    const long o_q1   = o_vd   + NN * 64;     // 4096
    const long o_q3   = o_q1   + 4096;        // 64
    const long o_z3   = o_q3   + 64;          // B*64
    const long o_op3  = o_z3   + (long)BGR * 64;  // B*64
    const long o_g1   = o_op3  + (long)BGR * 64;  // B*32
    const long o_g2   = o_g1   + (long)BGR * 32;  // B*16
    const long o_lg   = o_g2   + (long)BGR * 16;  // B*2
    // aliases
    const long o_scores = o_hW;               // <= B*64*256 == NN*64
    const long o_opre   = o_agg;              // B*64*64
    const long o_z1 = o_xcat;                 // 5 x B*64*64 fit in xcat region
    const long o_q2 = o_z1 + (long)BGR * 64 * 64;
    const long o_k2 = o_q2 + (long)BGR * 64 * 64;
    const long o_v2 = o_k2 + (long)BGR * 64 * 64;
    const long o_z2 = o_v2 + (long)BGR * 64 * 64;
    const long o_k3 = o_kd, o_v3 = o_vd;

    auto gguard = [&](const float* A, int lda, const float* W, int M, int K, int N,
                      const float* bias, float* C, int ldc, int relu) {
        dim3 g((M + 15) / 16, (N + 15) / 16);
        gemm_guard<<<g, 32, 0, stream>>>(A, lda, W, N, bias, C, ldc, M, K, N, relu);
    };
    auto fill0 = [&](float* p, long n) {
        k_fill0<<<(int)((n + 255) / 256), 256, 0, stream>>>(p, n);
    };

    // ---- degree / normalization -------------------------------------------
    fill0(ws + o_dis, NN);
    k_edge_deg<<<NEDGES / 256, 256, 0, stream>>>(dst, ws + o_dis, NEDGES);
    k_make_dis<<<(int)(NN / 256), 256, 0, stream>>>(ws + o_dis, (int)NN);

    // ---- GCN scatter helpers ----------------------------------------------
    auto gcn_edges32 = [&](const float* b, float* out, int ldo, int relu) {
        fill0(ws + o_agg, NN * 32);
        k_gcn_edge<32><<<(int)((long)NEDGES * 8 / 256), 256, 0, stream>>>(
            src, dst, ws + o_dis, ws + o_hW, ws + o_agg, NEDGES);
        k_gcn_combine<32><<<(int)(NN * 8 / 256), 256, 0, stream>>>(
            ws + o_hW, ws + o_agg, ws + o_dis, b, out, ldo, (int)NN, relu);
    };
    auto gcn_edges64 = [&](const float* b, float* out, int relu) {
        fill0(ws + o_agg, NN * 64);
        k_gcn_edge<64><<<(int)((long)NEDGES * 16 / 256), 256, 0, stream>>>(
            src, dst, ws + o_dis, ws + o_hW, ws + o_agg, NEDGES);
        k_gcn_combine<64><<<(int)(NN * 16 / 256), 256, 0, stream>>>(
            ws + o_hW, ws + o_agg, ws + o_dis, b, out, 64, (int)NN, relu);
    };

    // x1, x2, x3 -> xcat columns [0:32), [32:64), [64:96)
    gguard(x0, 3, conv1_w, (int)NN, 3, 32, nullptr, ws + o_hW, 32, 0);
    gcn_edges32(conv1_b, ws + o_xcat + 0, 96, 1);
    gemm_fast<2, 1><<<grid_tiles(NNODES), 32, 0, stream>>>(
        ws + o_xcat + 0, 96, conv2_w, nullptr, nullptr, 0, ws + o_hW, 32, NNODES, 0);
    gcn_edges32(conv2_b, ws + o_xcat + 32, 96, 1);
    gemm_fast<2, 1><<<grid_tiles(NNODES), 32, 0, stream>>>(
        ws + o_xcat + 32, 96, conv3_w, nullptr, nullptr, 0, ws + o_hW, 32, NNODES, 0);
    gcn_edges32(conv3_b, ws + o_xcat + 64, 96, 1);

    // h = xcat @ p_lin1 + b            [NN,64]
    gemm_fast<4, 3><<<grid_tiles(NNODES), 32, 0, stream>>>(
        ws + o_xcat, 96, plin1_w, plin1_b, nullptr, 0, ws + o_h, 64, NNODES, 0);

    // kd/vd = GCNConv(h) with mab1.k / mab1.v   [NN,64]
    gemm_fast<4, 2><<<grid_tiles(NNODES), 32, 0, stream>>>(
        ws + o_h, 64, m1_kw, nullptr, nullptr, 0, ws + o_hW, 64, NNODES, 0);
    gcn_edges64(m1_kb, ws + o_kd, 0);
    gemm_fast<4, 2><<<grid_tiles(NNODES), 32, 0, stream>>>(
        ws + o_h, 64, m1_vw, nullptr, nullptr, 0, ws + o_hW, 64, NNODES, 0);
    gcn_edges64(m1_vb, ws + o_vd, 0);

    // q1 = S1 @ mab1.q + b  (same for every graph)  [64,64]
    gemm_fast<4, 2><<<grid_tiles(64), 32, 0, stream>>>(
        S1, 64, m1_qw, m1_qb, nullptr, 0, ws + o_q1, 64, 64, 0);

    // ---- attention helper (per head) --------------------------------------
    auto mab_attn = [&](const float* q, long qstride, const float* K, const float* V,
                        float* opre, int S, int Sr, int Mk) {
        for (int head = 0; head < 4; ++head) {
            k_attn_scores<<<dim3(BGR, Sr / 16, Mk / 16), 32, 0, stream>>>(
                q, qstride, K, ws + o_scores, S, Sr, Mk, head);
            k_attn_softmax<<<dim3(BGR, S), 32, 0, stream>>>(ws + o_scores, Sr, Mk);
            k_attn_av<<<dim3(BGR, Sr / 16), 32, 0, stream>>>(
                ws + o_scores, V, q, qstride, opre, S, Sr, Mk, head);
        }
    };

    // ---- MAB1: GMPool_G  (S=64 seeds, M=256 keys) --------------------------
    mab_attn(ws + o_q1, 0, ws + o_kd, ws + o_vd, ws + o_opre, 64, 64, 256);
    gemm_fast<4, 2><<<grid_tiles(BGR * 64), 32, 0, stream>>>(
        ws + o_opre, 64, m1_ow, m1_ob, ws + o_opre, 64, ws + o_z1, 64, BGR * 64, 1);

    // ---- MAB2: SAB on z1  (S=64, M=64) ------------------------------------
    gemm_fast<4, 2><<<grid_tiles(BGR * 64), 32, 0, stream>>>(
        ws + o_z1, 64, m2_qw, m2_qb, nullptr, 0, ws + o_q2, 64, BGR * 64, 0);
    gemm_fast<4, 2><<<grid_tiles(BGR * 64), 32, 0, stream>>>(
        ws + o_z1, 64, m2_kw, m2_kb, nullptr, 0, ws + o_k2, 64, BGR * 64, 0);
    gemm_fast<4, 2><<<grid_tiles(BGR * 64), 32, 0, stream>>>(
        ws + o_z1, 64, m2_vw, m2_vb, nullptr, 0, ws + o_v2, 64, BGR * 64, 0);
    mab_attn(ws + o_q2, 64 * 64, ws + o_k2, ws + o_v2, ws + o_opre, 64, 64, 64);
    gemm_fast<4, 2><<<grid_tiles(BGR * 64), 32, 0, stream>>>(
        ws + o_opre, 64, m2_ow, m2_ob, ws + o_opre, 64, ws + o_z2, 64, BGR * 64, 1);

    // ---- MAB3: GMPool_I  (S=1, M=64) --------------------------------------
    gguard(S3, 64, m3_qw, 1, 64, 64, m3_qb, ws + o_q3, 64, 0);
    gemm_fast<4, 2><<<grid_tiles(BGR * 64), 32, 0, stream>>>(
        ws + o_z2, 64, m3_kw, m3_kb, nullptr, 0, ws + o_k3, 64, BGR * 64, 0);
    gemm_fast<4, 2><<<grid_tiles(BGR * 64), 32, 0, stream>>>(
        ws + o_z2, 64, m3_vw, m3_vb, nullptr, 0, ws + o_v3, 64, BGR * 64, 0);
    mab_attn(ws + o_q3, 0, ws + o_k3, ws + o_v3, ws + o_op3, 1, 16, 64);
    gemm_fast<4, 2><<<grid_tiles(BGR), 32, 0, stream>>>(
        ws + o_op3, 64, m3_ow, m3_ob, ws + o_op3, 64, ws + o_z3, 64, BGR, 1);

    // ---- head --------------------------------------------------------------
    gemm_fast<2, 2><<<grid_tiles(BGR), 32, 0, stream>>>(
        ws + o_z3, 64, plin2_w, plin2_b, nullptr, 0, ws + o_g1, 32, BGR, 0);
    gemm_fast<1, 1><<<grid_tiles(BGR), 32, 0, stream>>>(
        ws + o_g1, 32, h1_w, h1_b, nullptr, 0, ws + o_g2, 16, BGR, 1);
    gguard(ws + o_g2, 16, h2_w, BGR, 16, 2, h2_b, ws + o_lg, 2, 0);
    k_log_softmax2<<<(BGR + 255) / 256, 256, 0, stream>>>(ws + o_lg, (float*)d_out, BGR);
}
